// transformer_block_22256520528887
// MI455X (gfx1250) — compile-verified, matrix operand order checked by reading, not run on hardware
//
#include <hip/hip_runtime.h>
#include <hip/hip_bf16.h>

// ---------------------------------------------------------------------------
// MI455X (gfx1250) transformer block.
// Compute-bound (~137 GFLOP vs ~µs of HBM traffic at 23.3 TB/s): every matmul
// runs on v_wmma_f32_16x16x32_bf16 (wave32, f32 accumulate), and GEMM tiles
// are streamed into LDS with the Tensor Data Mover (tensor_load_to_lds,
// TENSORcnt double-buffered) so global->LDS traffic overlaps the WMMAs.
// ---------------------------------------------------------------------------

#define TB_D_MODEL 1024
#define TB_N_HEADS 16
#define TB_D_K     64
#define TB_D_FF    2688
#define TB_B       2
#define TB_S       2048

typedef __bf16 bf16_t;
typedef __attribute__((ext_vector_type(16))) __bf16 v16bf;
typedef __attribute__((ext_vector_type(8)))  float  v8f;
typedef __attribute__((ext_vector_type(4)))  unsigned int u32x4;
typedef __attribute__((ext_vector_type(8)))  int i32x8;
typedef __attribute__((ext_vector_type(4)))  int i32x4;

union Frag {           // one WMMA 16x16x32 bf16 operand (per-lane 16 bf16)
  v16bf v;
  uint4 u[2];
};
union Pack8 {
  uint4  u;
  bf16_t e[8];
};

static __device__ __forceinline__ v8f wmma_bf16(v16bf a, v16bf b, v8f c) {
  return __builtin_amdgcn_wmma_f32_16x16x32_bf16(false, a, false, b,
                                                 (short)0, c, false, false);
}

// ---------------------------------------------------------------------------
// TDM: 2D tile load, global -> LDS.  D# per CDNA5 ISA ch.8:
//  group0: [1:0]count=1, [63:32]lds_addr, [120:64]global_addr, [127:126]type=2
//  group1: [17:16]data_size=1(2B), tensor_dim0/1, tile_dim0/1, dim0_stride
//  groups 2/3 (+ extra operand): zero (tile_dim2..4 = 0 -> unused, 2D tile)
// This toolchain's builtin arity is 6 (clang-23 / therock-10.0 headers).
// ---------------------------------------------------------------------------
static __device__ __forceinline__ void tdm_load_tile_bf16(
    const bf16_t* gptr, unsigned int ldsByteOff,
    unsigned int tensorD0, unsigned int tensorD1, unsigned int strideD0,
    unsigned int tileW, unsigned int tileH) {
  const unsigned long long ga = (unsigned long long)(uintptr_t)gptr;
  u32x4 g0;
  g0[0] = 1u;                                            // count=1, user mode
  g0[1] = ldsByteOff;                                    // lds_addr (bytes)
  g0[2] = (unsigned int)ga;                              // global_addr[31:0]
  g0[3] = (unsigned int)((ga >> 32) & 0x1FFFFFFu)        // global_addr[56:32]
        | (2u << 30);                                    // type=2 ("image")
  i32x8 g1;
  g1[0] = (int)(1u << 16);                               // data_size=1 -> 2B
  g1[1] = (int)((tensorD0 & 0xFFFFu) << 16);             // tensor_dim0[15:0]
  g1[2] = (int)((tensorD0 >> 16) | ((tensorD1 & 0xFFFFu) << 16));
  g1[3] = (int)((tensorD1 >> 16) | (tileW << 16));       // tile_dim0
  g1[4] = (int)tileH;                                    // tile_dim1 (dim2=0)
  g1[5] = (int)strideD0;                                 // dim0_stride[31:0]
  g1[6] = 0;                                             // stride hi, dim1_stride lo
  g1[7] = 0;
  const i32x4 z4 = {0, 0, 0, 0};
  const i32x8 z8 = {0, 0, 0, 0, 0, 0, 0, 0};
  __builtin_amdgcn_tensor_load_to_lds(g0, g1, z4, z4, z8, 0);
}

// ---------------------------------------------------------------------------
// f32 -> bf16 conversion (weights)
// ---------------------------------------------------------------------------
__global__ void k_f32_to_bf16(const float* __restrict__ in,
                              bf16_t* __restrict__ out, size_t n) {
  size_t i = (size_t)blockIdx.x * 256 + threadIdx.x;
  if (i < n) out[i] = (bf16_t)in[i];
}

// ---------------------------------------------------------------------------
// RMSNorm: one 256-thread block per row of D=1024; bf16 output for WMMA GEMMs
// ---------------------------------------------------------------------------
__global__ __launch_bounds__(256)
void k_rmsnorm(const float* __restrict__ x, const float* __restrict__ w,
               bf16_t* __restrict__ out) {
  __shared__ float red[8];
  const int row = blockIdx.x;
  const float* xr = x + (size_t)row * TB_D_MODEL;
  float ss = 0.f;
  #pragma unroll
  for (int i = 0; i < TB_D_MODEL / 256; ++i) {
    float v = xr[threadIdx.x + i * 256];
    ss += v * v;
  }
  #pragma unroll
  for (int m = 1; m < 32; m <<= 1) ss += __shfl_xor(ss, m, 32);
  if ((threadIdx.x & 31) == 0) red[threadIdx.x >> 5] = ss;
  __syncthreads();
  float tot = 0.f;
  #pragma unroll
  for (int i = 0; i < 8; ++i) tot += red[i];
  const float r = rsqrtf(tot / (float)TB_D_MODEL + 1e-5f);
  #pragma unroll
  for (int i = 0; i < TB_D_MODEL / 256; ++i) {
    int c = threadIdx.x + i * 256;
    out[(size_t)row * TB_D_MODEL + c] = (bf16_t)(xr[c] * w[c] * r);
  }
}

// ---------------------------------------------------------------------------
// WMMA GEMM:  C[M,N] = A[M,K] * W[N,K]^T
// 256 threads = 8 waves; block tile 128x128, K tile 64.
// Dynamic LDS: [A0|B0|A1|B1], 16KB each (64KB). TDM double buffering:
// wave 0 streams tile i+1 with tensor_load_to_lds while all waves WMMA on
// tile i; s_wait_tensorcnt(0) + workgroup barrier hands buffers over.
// outMode 0: bf16 C      outMode 1: f32 C = Res + acc (residual fused)
// ---------------------------------------------------------------------------
#define GBM 128
#define GBN 128
#define GBK 64
#define GTILE (GBM * GBK * 2)          // 16384 bytes per tile buffer

__global__ __launch_bounds__(256)
void k_gemm_wmma(const bf16_t* __restrict__ A, const bf16_t* __restrict__ W,
                 void* __restrict__ Cout, const float* __restrict__ Res,
                 int M, int N, int K, int outMode) {
  extern __shared__ char smem[];       // 4 * GTILE bytes
  const unsigned int ldsBase = __builtin_amdgcn_groupstaticsize();

  const int tid  = threadIdx.x;
  const int wid  = tid >> 5;
  const int lane = tid & 31;
  const int half = lane >> 4;
  const int r    = lane & 15;
  const int waveM = (wid & 1) * 64;
  const int waveN = (wid >> 1) * 32;
  const int mBase = blockIdx.y * GBM;
  const int nBase = blockIdx.x * GBN;

  v8f acc[4][2];
  #pragma unroll
  for (int mt = 0; mt < 4; ++mt)
    #pragma unroll
    for (int nt = 0; nt < 2; ++nt) acc[mt][nt] = (v8f){};

  // prologue: stream K-tile 0 into buffer 0
  if (wid == 0) {
    tdm_load_tile_bf16(A + (size_t)mBase * K, ldsBase,          K, GBM, K, GBK, GBM);
    tdm_load_tile_bf16(W + (size_t)nBase * K, ldsBase + GTILE,  K, GBN, K, GBK, GBN);
  }

  int buf = 0;
  for (int k0 = 0; k0 < K; k0 += GBK) {
    if (wid == 0) __builtin_amdgcn_s_wait_tensorcnt(0);  // buf ready (wave 0)
    __syncthreads();                                     // release consumers;
                                                         // also: buf^1 readers
                                                         // from iter i-1 done
    if (wid == 0 && (k0 + GBK) < K) {                    // stream next tile
      const unsigned int nb = ldsBase + (unsigned int)(buf ^ 1) * (2 * GTILE);
      tdm_load_tile_bf16(A + (size_t)mBase * K + k0 + GBK, nb,          K, GBM, K, GBK, GBM);
      tdm_load_tile_bf16(W + (size_t)nBase * K + k0 + GBK, nb + GTILE,  K, GBN, K, GBK, GBN);
    }

    const bf16_t* As = (const bf16_t*)(smem + buf * (2 * GTILE));
    const bf16_t* Bs = (const bf16_t*)(smem + buf * (2 * GTILE) + GTILE);

    #pragma unroll
    for (int ks = 0; ks < 2; ++ks) {
      const int kc0 = ks * 32 + half * 8;
      const int kc1 = ks * 32 + 16 + half * 8;
      Frag fa[4], fb[2];
      #pragma unroll
      for (int mt = 0; mt < 4; ++mt) {
        const bf16_t* p = As + (waveM + mt * 16 + r) * GBK;
        fa[mt].u[0] = *(const uint4*)(p + kc0);
        fa[mt].u[1] = *(const uint4*)(p + kc1);
      }
      #pragma unroll
      for (int nt = 0; nt < 2; ++nt) {
        const bf16_t* p = Bs + (waveN + nt * 16 + r) * GBK;
        fb[nt].u[0] = *(const uint4*)(p + kc0);
        fb[nt].u[1] = *(const uint4*)(p + kc1);
      }
      #pragma unroll
      for (int mt = 0; mt < 4; ++mt)
        #pragma unroll
        for (int nt = 0; nt < 2; ++nt)
          acc[mt][nt] = wmma_bf16(fa[mt].v, fb[nt].v, acc[mt][nt]);
    }
    buf ^= 1;
  }

  // epilogue: C/D layout -> row = tile + half*8 + j, col = tile + r
  #pragma unroll
  for (int mt = 0; mt < 4; ++mt)
    #pragma unroll
    for (int nt = 0; nt < 2; ++nt)
      #pragma unroll
      for (int j = 0; j < 8; ++j) {
        const int row = mBase + waveM + mt * 16 + half * 8 + j;
        const int col = nBase + waveN + nt * 16 + r;
        const size_t off = (size_t)row * N + col;
        const float v = acc[mt][nt][j];
        if (outMode == 0) ((bf16_t*)Cout)[off] = (bf16_t)v;
        else              ((float*)Cout)[off]  = Res[off] + v;
      }
}

// ---------------------------------------------------------------------------
// RoPE (bf16 in/out), layout (b, s, h, d), pairs (2k, 2k+1)
// ---------------------------------------------------------------------------
__global__ void k_rope(const bf16_t* __restrict__ in, bf16_t* __restrict__ out,
                       const int* __restrict__ pos, size_t n) {
  size_t i = (size_t)blockIdx.x * 256 + threadIdx.x;
  if (i >= n) return;
  const int halfdk = TB_D_K / 2;
  const int k = (int)(i % halfdk);
  size_t rest = i / halfdk;
  const int h = (int)(rest % TB_N_HEADS);
  const size_t bs = rest / TB_N_HEADS;         // b*S + s
  const int s = (int)(bs % TB_S);
  const float p = (float)pos[s];
  const float inv = __powf(10000.0f, -2.0f * (float)k / (float)TB_D_K);
  float sn, cs;
  __sincosf(p * inv, &sn, &cs);
  const size_t base = (bs * TB_N_HEADS + h) * TB_D_K + 2 * k;
  const float xe = (float)in[base], xo = (float)in[base + 1];
  out[base]     = (bf16_t)(xe * cs - xo * sn);
  out[base + 1] = (bf16_t)(xe * sn + xo * cs);
}

// ---------------------------------------------------------------------------
// Causal flash attention, WMMA for QK^T and PV.
// 128 threads = 4 waves. Q tile 64 rows (16 rows/wave), KV tile 128.
// Static LDS = 8K(Q) + 16K(K) + 16K(V^T) + 16K(P) = 56 KB.
// ---------------------------------------------------------------------------
__global__ __launch_bounds__(128)
void k_attn(const bf16_t* __restrict__ Qr, const bf16_t* __restrict__ Kr,
            const bf16_t* __restrict__ Vb, bf16_t* __restrict__ ctx) {
  __shared__ __align__(16) bf16_t Qs[64 * 64];
  __shared__ __align__(16) bf16_t Ks[128 * 64];
  __shared__ __align__(16) bf16_t Vt[64 * 128];   // [d][kv]
  __shared__ __align__(16) bf16_t Ps[4][16 * 128];

  const int qt = blockIdx.x;       // q tile (64 rows)
  const int h  = blockIdx.y;
  const int b  = blockIdx.z;
  const int tid  = threadIdx.x;
  const int wid  = tid >> 5;
  const int lane = tid & 31;
  const int half = lane >> 4;
  const int r    = lane & 15;

  // load Q tile: 64 rows x cols 0..31 (256 uint4)
  for (int i = tid; i < 256; i += 128) {
    const int rr = i >> 2, cc = i & 3;  // 4 uint4 (cols 0..31) per row, packed 32-wide in Qs
    const size_t g = ((size_t)(b * TB_S + qt * 64 + rr) * TB_N_HEADS + h) * TB_D_K + cc * 8;
    ((uint4*)Qs)[i] = *(const uint4*)(Qr + g);
  }

  v8f ctxacc[4];
  #pragma unroll
  for (int dt = 0; dt < 4; ++dt) ctxacc[dt] = (v8f){};
  float mrow[8], lrow[8];
  #pragma unroll
  for (int j = 0; j < 8; ++j) { mrow[j] = -1e30f; lrow[j] = 0.f; }

  const float scale = 0.125f;                    // 1/sqrt(64)
  const int qrow0 = qt * 64 + wid * 16 + half * 8;
  const int ktmax = ((qt + 1) * 64 - 1) >> 7;

  for (int kt = 0; kt <= ktmax; ++kt) {
    __syncthreads();   // prior-iteration Ks/Vt readers done
    for (int i = tid; i < 512; i += 128) {
      const int rr = i >> 3, cc = i & 7;
      const size_t g = ((size_t)(b * TB_S + kt * 128 + rr) * TB_N_HEADS + h) * TB_D_K + cc * 8;
      ((uint4*)Ks)[i] = *(const uint4*)(Kr + g);
      Pack8 pv; pv.u = *(const uint4*)(Vb + g);
      #pragma unroll
      for (int e = 0; e < 8; ++e) Vt[(cc * 8 + e) * 128 + rr] = pv.e[e];
    }
    __syncthreads();

    // scores: 16 q rows x 128 kv cols per wave
    v8f sacc[8];
    #pragma unroll
    for (int ct = 0; ct < 8; ++ct) sacc[ct] = (v8f){};
    #pragma unroll
    for (int ks = 0; ks < 2; ++ks) {
      const int kc0 = ks * 32 + half * 8;
      const int kc1 = ks * 32 + 16 + half * 8;
      Frag fa;
      const bf16_t* ap = Qs + (wid * 16 + r) * 64;
      fa.u[0] = *(const uint4*)(ap + kc0);
      fa.u[1] = *(const uint4*)(ap + kc1);
      #pragma unroll
      for (int ct = 0; ct < 8; ++ct) {
        Frag fb;
        const bf16_t* bp = Ks + (ct * 16 + r) * 64;
        fb.u[0] = *(const uint4*)(bp + kc0);
        fb.u[1] = *(const uint4*)(bp + kc1);
        sacc[ct] = wmma_bf16(fa.v, fb.v, sacc[ct]);
      }
    }

    // online softmax per owned row j (row = qrow0 + j; cols striped over lanes)
    #pragma unroll
    for (int j = 0; j < 8; ++j) {
      const int qg = qrow0 + j;
      float sv[8], mx = -1e30f;
      #pragma unroll
      for (int ct = 0; ct < 8; ++ct) {
        float s = sacc[ct][j] * scale;
        const int col = kt * 128 + ct * 16 + r;
        if (col > qg) s = -10000.0f;            // causal mask (ref semantics)
        sv[ct] = s;
        mx = fmaxf(mx, s);
      }
      #pragma unroll
      for (int m = 1; m < 16; m <<= 1) mx = fmaxf(mx, __shfl_xor(mx, m, 32));
      const float mnew = fmaxf(mrow[j], mx);
      const float corr = __expf(mrow[j] - mnew);
      float rs = 0.f;
      #pragma unroll
      for (int ct = 0; ct < 8; ++ct) {
        const float p = __expf(sv[ct] - mnew);
        rs += p;
        Ps[wid][(half * 8 + j) * 128 + ct * 16 + r] = (bf16_t)p;
      }
      #pragma unroll
      for (int m = 1; m < 16; m <<= 1) rs += __shfl_xor(rs, m, 32);
      lrow[j] = lrow[j] * corr + rs;
      mrow[j] = mnew;
      #pragma unroll
      for (int dt = 0; dt < 4; ++dt) ctxacc[dt][j] *= corr;
    }

    // ctx += P(16x128) @ V(128x64); P read back from wave-private LDS strip
    #pragma unroll
    for (int ks = 0; ks < 4; ++ks) {
      const int kc0 = ks * 32 + half * 8;
      const int kc1 = ks * 32 + 16 + half * 8;
      Frag fa;
      const bf16_t* ap = &Ps[wid][r * 128];
      fa.u[0] = *(const uint4*)(ap + kc0);
      fa.u[1] = *(const uint4*)(ap + kc1);
      #pragma unroll
      for (int dt = 0; dt < 4; ++dt) {
        Frag fb;
        const bf16_t* bp = Vt + (dt * 16 + r) * 128;
        fb.u[0] = *(const uint4*)(bp + kc0);
        fb.u[1] = *(const uint4*)(bp + kc1);
        ctxacc[dt] = wmma_bf16(fa.v, fb.v, ctxacc[dt]);
      }
    }
  }

  // write ctx (b, s, h, d) in bf16 for the O-projection GEMM
  #pragma unroll
  for (int dt = 0; dt < 4; ++dt)
    #pragma unroll
    for (int j = 0; j < 8; ++j) {
      const int qg = qrow0 + j;
      const int d  = dt * 16 + r;
      ctx[((size_t)(b * TB_S + qg) * TB_N_HEADS + h) * TB_D_K + d] =
          (bf16_t)(ctxacc[dt][j] / lrow[j]);
    }
}

// ---------------------------------------------------------------------------
// SwiGLU gate: out = silu(g) * u
// ---------------------------------------------------------------------------
__global__ void k_silu_mul(const bf16_t* __restrict__ g,
                           const bf16_t* __restrict__ u,
                           bf16_t* __restrict__ out, size_t n) {
  size_t i = (size_t)blockIdx.x * 256 + threadIdx.x;
  if (i < n) {
    const float gf = (float)g[i];
    const float uf = (float)u[i];
    out[i] = (bf16_t)((gf / (1.f + __expf(-gf))) * uf);
  }
}

// ---------------------------------------------------------------------------
// Host orchestration
// ---------------------------------------------------------------------------
extern "C" void kernel_launch(void* const* d_in, const int* in_sizes, int n_in,
                              void* d_out, int out_size, void* d_ws, size_t ws_size,
                              hipStream_t stream) {
  (void)in_sizes; (void)n_in; (void)out_size; (void)ws_size;
  const int M = TB_B * TB_S;             // 4096
  const int D = TB_D_MODEL;              // 1024
  const int F = TB_D_FF;                 // 2688
  const unsigned GEMM_LDS = 4 * GTILE;   // 64 KB dynamic LDS (double buffer)

  const float* x    = (const float*)d_in[0];
  const float* ln1w = (const float*)d_in[1];
  const float* qw   = (const float*)d_in[2];
  const float* kw   = (const float*)d_in[3];
  const float* vw   = (const float*)d_in[4];
  const float* ow   = (const float*)d_in[5];
  const float* ln2w = (const float*)d_in[6];
  const float* w1w  = (const float*)d_in[7];
  const float* w3w  = (const float*)d_in[8];
  const float* w2w  = (const float*)d_in[9];
  const int*   pos  = (const int*)d_in[10];
  float* out = (float*)d_out;

  char* ws = (char*)d_ws;
  size_t off = 0;
  auto alloc = [&](size_t bytes) -> void* {
    void* p = ws + off;
    off += (bytes + 255) & ~(size_t)255;
    return p;
  };

  bf16_t* hb  = (bf16_t*)alloc((size_t)M * D * 2);
  bf16_t* wqb = (bf16_t*)alloc((size_t)D * D * 2);
  bf16_t* wkb = (bf16_t*)alloc((size_t)D * D * 2);
  bf16_t* wvb = (bf16_t*)alloc((size_t)D * D * 2);
  bf16_t* wob = (bf16_t*)alloc((size_t)D * D * 2);
  bf16_t* w1b = (bf16_t*)alloc((size_t)F * D * 2);
  bf16_t* w3b = (bf16_t*)alloc((size_t)F * D * 2);
  bf16_t* w2b = (bf16_t*)alloc((size_t)D * F * 2);
  bf16_t* qb  = (bf16_t*)alloc((size_t)M * D * 2);
  bf16_t* kb  = (bf16_t*)alloc((size_t)M * D * 2);
  bf16_t* vb  = (bf16_t*)alloc((size_t)M * D * 2);
  bf16_t* qr  = (bf16_t*)alloc((size_t)M * D * 2);
  bf16_t* kr  = (bf16_t*)alloc((size_t)M * D * 2);
  bf16_t* ctb = (bf16_t*)alloc((size_t)M * D * 2);
  float*  x1  = (float*) alloc((size_t)M * D * 4);
  bf16_t* h2b = (bf16_t*)alloc((size_t)M * D * 2);
  bf16_t* gb  = (bf16_t*)alloc((size_t)M * F * 2);
  bf16_t* ub  = (bf16_t*)alloc((size_t)M * F * 2);
  bf16_t* fb  = (bf16_t*)alloc((size_t)M * F * 2);

  auto cvt = [&](const float* src, bf16_t* dst, size_t n) {
    k_f32_to_bf16<<<(unsigned)((n + 255) / 256), 256, 0, stream>>>(src, dst, n);
  };
  cvt(qw, wqb, (size_t)D * D);
  cvt(kw, wkb, (size_t)D * D);
  cvt(vw, wvb, (size_t)D * D);
  cvt(ow, wob, (size_t)D * D);
  cvt(w1w, w1b, (size_t)F * D);
  cvt(w3w, w3b, (size_t)F * D);
  cvt(w2w, w2b, (size_t)D * F);

  // h = rmsnorm(x, ln1)
  k_rmsnorm<<<M, 256, 0, stream>>>(x, ln1w, hb);

  // Q/K/V projections (bf16 outputs)
  dim3 gQKV(D / GBN, M / GBM);
  k_gemm_wmma<<<gQKV, 256, GEMM_LDS, stream>>>(hb, wqb, qb, nullptr, M, D, D, 0);
  k_gemm_wmma<<<gQKV, 256, GEMM_LDS, stream>>>(hb, wkb, kb, nullptr, M, D, D, 0);
  k_gemm_wmma<<<gQKV, 256, GEMM_LDS, stream>>>(hb, wvb, vb, nullptr, M, D, D, 0);

  // RoPE on Q and K
  const size_t nrope = (size_t)M * TB_N_HEADS * (TB_D_K / 2);
  k_rope<<<(unsigned)((nrope + 255) / 256), 256, 0, stream>>>(qb, qr, pos, nrope);
  k_rope<<<(unsigned)((nrope + 255) / 256), 256, 0, stream>>>(kb, kr, pos, nrope);

  // causal flash attention -> ctx (bf16)
  dim3 gAttn(TB_S / 64, TB_N_HEADS, TB_B);
  k_attn<<<gAttn, 128, 0, stream>>>(qr, kr, vb, ctb);

  // x1 = x + ctx @ o_w^T  (f32)
  k_gemm_wmma<<<gQKV, 256, GEMM_LDS, stream>>>(ctb, wob, x1, x, M, D, D, 1);

  // h2 = rmsnorm(x1, ln2)
  k_rmsnorm<<<M, 256, 0, stream>>>(x1, ln2w, h2b);

  // g = h2 @ w1^T, u = h2 @ w3^T  (bf16)
  dim3 gFF(F / GBN, M / GBM);
  k_gemm_wmma<<<gFF, 256, GEMM_LDS, stream>>>(h2b, w1b, gb, nullptr, M, F, D, 0);
  k_gemm_wmma<<<gFF, 256, GEMM_LDS, stream>>>(h2b, w3b, ub, nullptr, M, F, D, 0);

  // ffn_in = silu(g) * u
  const size_t nff = (size_t)M * F;
  k_silu_mul<<<(unsigned)((nff + 255) / 256), 256, 0, stream>>>(gb, ub, fb, nff);

  // out = x1 + ffn_in @ w2^T  (f32, final output)
  dim3 gW2(D / GBN, M / GBM);
  k_gemm_wmma<<<gW2, 256, GEMM_LDS, stream>>>(fb, w2b, out, x1, M, D, F, 1);
}
